// SelfAttention_11974368821674
// MI455X (gfx1250) — matrix operand order repaired and checked by reading, not compiled, not run on hardware
//
#include <hip/hip_runtime.h>

// ---------------------------------------------------------------------------
// Problem constants
// ---------------------------------------------------------------------------
static constexpr int Lq   = 2048;
static constexpr int DIM  = 1536;
static constexpr int H    = 24;
static constexpr int HD   = 64;          // head dim
static constexpr int NQKV = 3 * DIM;     // 4608
static constexpr float EPS = 1e-6f;

typedef __attribute__((ext_vector_type(16))) _Float16 v16h;
typedef __attribute__((ext_vector_type(8)))  _Float16 v8h;
typedef __attribute__((ext_vector_type(8)))  float    v8f;

// Build a 16-half fragment from two 16-byte chunks.
__device__ __forceinline__ v16h ldfrag(const _Float16* lo, const _Float16* hi) {
    v8h a = *(const v8h*)lo;
    v8h b = *(const v8h*)hi;
    return __builtin_shufflevector(a, b, 0,1,2,3,4,5,6,7,8,9,10,11,12,13,14,15);
}

__device__ __forceinline__ v8f wmma16(v16h a, v16h b, v8f c) {
    // D = A(16x32 f16) * B(32x16 f16) + C(16x16 f32)
    return __builtin_amdgcn_wmma_f32_16x16x32_f16(
        /*neg_a=*/false, a, /*neg_b=*/false, b,
        /*c_mod=*/(short)0, c, /*reuse_a=*/false, /*reuse_b=*/false);
}

// ---------------------------------------------------------------------------
// 1) fp32 -> fp16 conversion (grid-stride)
// ---------------------------------------------------------------------------
__global__ __launch_bounds__(256)
void cvt_f32_to_f16(const float* __restrict__ src, _Float16* __restrict__ dst, int n) {
    int i = blockIdx.x * blockDim.x + threadIdx.x;
    int stride = gridDim.x * blockDim.x;
    for (; i < n; i += stride) dst[i] = (_Float16)src[i];
}

// ---------------------------------------------------------------------------
// 2/5) WMMA GEMM: C[M][N] = A[M][K] * Bw[N][K]^T + bias[N]   (f16 in, f32 out)
//      One wave computes a 64x64 output block (4x4 tiles of 16x16).
//      B fragments are loaded once per K-step and reused across 4 A rows:
//      16 b128 loads feed 16 WMMAs (1.0 wmma/load vs 0.5 for 32x32 tiles).
// ---------------------------------------------------------------------------
__global__ __launch_bounds__(256)
void wmma_gemm_f16(const _Float16* __restrict__ A,
                   const _Float16* __restrict__ Bw,
                   const float* __restrict__ bias,
                   float* __restrict__ C,
                   int M, int N, int K) {
    int wid    = (blockIdx.x * blockDim.x + threadIdx.x) >> 5;
    int lane   = threadIdx.x & 31;
    int tilesM = M >> 6;               // 64-row blocks
    int wm     = wid % tilesM;
    int wn     = wid / tilesM;
    if (wn >= (N >> 6)) return;

    int m0 = wm * 64, n0 = wn * 64;
    int l16 = lane & 15;
    int hh  = (lane >> 4) & 1;         // which 16-lane half

    v8f acc[4][4] = {};

    const _Float16* abase[4];
    const _Float16* bbase[4];
    #pragma unroll
    for (int i = 0; i < 4; ++i)
        abase[i] = A + (size_t)(m0 + i * 16 + l16) * K + hh * 8;
    #pragma unroll
    for (int j = 0; j < 4; ++j)
        bbase[j] = Bw + (size_t)(n0 + j * 16 + l16) * K + hh * 8;

    for (int k0 = 0; k0 < K; k0 += 32) {
        // B fragments: lane n holds column n (a row of Bw); K split by half.
        v16h bf[4];
        #pragma unroll
        for (int j = 0; j < 4; ++j)
            bf[j] = ldfrag(bbase[j] + k0, bbase[j] + k0 + 16);

        // Prefetch next K-slab (emits global_prefetch_b8).
        __builtin_prefetch(bbase[0] + k0 + 32, 0, 3);
        __builtin_prefetch(abase[0] + k0 + 32, 0, 3);

        // A fragments: lane m holds row m; halves split K per ISA layout.
        #pragma unroll
        for (int i = 0; i < 4; ++i) {
            v16h af = ldfrag(abase[i] + k0, abase[i] + k0 + 16);
            #pragma unroll
            for (int j = 0; j < 4; ++j)
                acc[i][j] = wmma16(af, bf[j], acc[i][j]);
        }
    }

    #pragma unroll
    for (int j = 0; j < 4; ++j) {
        int   n  = n0 + j * 16 + l16;
        float bv = bias[n];
        #pragma unroll
        for (int i = 0; i < 4; ++i) {
            int mrow = m0 + i * 16 + hh * 8;
            #pragma unroll
            for (int r = 0; r < 8; ++r)
                C[(size_t)(mrow + r) * N + n] = acc[i][j][r] + bv;
        }
    }
}

// ---------------------------------------------------------------------------
// 3) Per-(token, head): RMSNorm q,k -> f16; V -> f16 transposed per head.
//    One wave per (token, head); 8 independent waves per block.
// ---------------------------------------------------------------------------
__global__ __launch_bounds__(256)
void rmsnorm_qkv(const float* __restrict__ qkv,
                 const float* __restrict__ g_q,
                 const float* __restrict__ g_k,
                 _Float16* __restrict__ qh,
                 _Float16* __restrict__ kh,
                 _Float16* __restrict__ vt) {
    int pair = blockIdx.x * 8 + (threadIdx.x >> 5);
    int lane = threadIdx.x & 31;
    int t = pair / H;
    int h = pair % H;

    const float* qrow = qkv + (size_t)t * NQKV + h * HD;
    const float* krow = qrow + DIM;
    const float* vrow = qrow + 2 * DIM;

    float q0 = qrow[lane], q1 = qrow[lane + 32];
    float k0 = krow[lane], k1 = krow[lane + 32];

    float sq = q0 * q0 + q1 * q1;
    float sk = k0 * k0 + k1 * k1;
    for (int m = 16; m >= 1; m >>= 1) {
        sq += __shfl_xor(sq, m, 32);
        sk += __shfl_xor(sk, m, 32);
    }
    float rq = rsqrtf(sq * (1.0f / HD) + EPS);
    float rk = rsqrtf(sk * (1.0f / HD) + EPS);

    _Float16* qo = qh + (size_t)t * DIM + h * HD;
    _Float16* ko = kh + (size_t)t * DIM + h * HD;
    qo[lane]      = (_Float16)(q0 * rq * g_q[lane]);
    qo[lane + 32] = (_Float16)(q1 * rq * g_q[lane + 32]);
    ko[lane]      = (_Float16)(k0 * rk * g_k[lane]);
    ko[lane + 32] = (_Float16)(k1 * rk * g_k[lane + 32]);

    // vt[h][d][t] : contiguous along tokens for PV B-operand loads
    _Float16* vo = vt + (size_t)h * HD * Lq + t;
    vo[(size_t)lane * Lq]        = (_Float16)vrow[lane];
    vo[(size_t)(lane + 32) * Lq] = (_Float16)vrow[lane + 32];
}

// ---------------------------------------------------------------------------
// 4) Flash attention: one wave per (head, 32-query tile) = two 16-row WMMA
//    tiles sharing every K/V fragment (halves L2 K/V traffic vs 16 rows).
//    4 waves per workgroup, each with its own LDS P-buffer; all waves run
//    identical trip counts so __syncthreads() is uniform.
// ---------------------------------------------------------------------------
__global__ __launch_bounds__(128)
void flash_attn(const _Float16* __restrict__ qh,
                const _Float16* __restrict__ kh,
                const _Float16* __restrict__ vt,
                _Float16* __restrict__ oh) {
    __shared__ _Float16 sP[4][32][32];   // [wave][query row][key in 32-chunk]

    int wib = threadIdx.x >> 5;          // wave index in block
    int gw  = blockIdx.x * 4 + wib;      // global wave id
    int h   = gw % H;
    int qt  = gw / H;                    // 32-query tile index
    int qr0 = qt * 32;

    int lane = threadIdx.x & 31;
    int l16  = lane & 15;
    int hh   = (lane >> 4) & 1;

    // Q fragments for both 16-row tiles (head dim K=64 -> two 32-wide frags).
    v16h qa[2][2];
    #pragma unroll
    for (int t = 0; t < 2; ++t) {
        const _Float16* qbase =
            qh + (size_t)(qr0 + t * 16 + l16) * DIM + h * HD + hh * 8;
        qa[t][0] = ldfrag(qbase,      qbase + 16);
        qa[t][1] = ldfrag(qbase + 32, qbase + 48);
    }

    v8f o[2][4] = {};
    float mst[2][8], lst[2][8];
    #pragma unroll
    for (int t = 0; t < 2; ++t)
        for (int r = 0; r < 8; ++r) { mst[t][r] = -3.0e38f; lst[t][r] = 0.0f; }
    const float scale = 0.125f;          // 1/sqrt(64)

    const _Float16* vhead = vt + (size_t)h * HD * Lq;

    for (int kt = 0; kt < Lq; kt += 32) {
        // ---- K fragments for 32 keys, shared by both q-tiles ----
        const _Float16* kb0 = kh + (size_t)(kt + l16) * DIM + h * HD + hh * 8;
        const _Float16* kb1 = kb0 + (size_t)16 * DIM;
        v16h b00 = ldfrag(kb0,      kb0 + 16);   // keys kt..+16, dims 0..31
        v16h b01 = ldfrag(kb0 + 32, kb0 + 48);   // keys kt..+16, dims 32..63
        v16h b10 = ldfrag(kb1,      kb1 + 16);
        v16h b11 = ldfrag(kb1 + 32, kb1 + 48);

        __syncthreads();                 // WAR: previous iteration's sP reads
        #pragma unroll
        for (int t = 0; t < 2; ++t) {
            v8f s0 = {}, s1 = {};
            s0 = wmma16(qa[t][0], b00, s0);
            s0 = wmma16(qa[t][1], b01, s0);
            s1 = wmma16(qa[t][0], b10, s1);
            s1 = wmma16(qa[t][1], b11, s1);

            // ---- online softmax over the 32 new keys ----
            float al[8];
            #pragma unroll
            for (int r = 0; r < 8; ++r) {
                float v0 = s0[r] * scale, v1 = s1[r] * scale;
                float mx = fmaxf(v0, v1);
                for (int m = 1; m < 16; m <<= 1)
                    mx = fmaxf(mx, __shfl_xor(mx, m, 32));
                float nm = fmaxf(mst[t][r], mx);
                float p0 = __expf(v0 - nm);
                float p1 = __expf(v1 - nm);
                float su = p0 + p1;
                for (int m = 1; m < 16; m <<= 1)
                    su += __shfl_xor(su, m, 32);
                al[r]     = __expf(mst[t][r] - nm);
                lst[t][r] = lst[t][r] * al[r] + su;
                mst[t][r] = nm;
                // C/D layout -> LDS (rows r / r+8 across halves)
                sP[wib][t * 16 + hh * 8 + r][l16]      = (_Float16)p0;
                sP[wib][t * 16 + hh * 8 + r][16 + l16] = (_Float16)p1;
            }
            #pragma unroll
            for (int r = 0; r < 8; ++r) {
                o[t][0][r] *= al[r]; o[t][1][r] *= al[r];
                o[t][2][r] *= al[r]; o[t][3][r] *= al[r];
            }
        }
        __syncthreads();

        // ---- P fragments (A layout) from LDS ----
        v16h pa[2];
        #pragma unroll
        for (int t = 0; t < 2; ++t) {
            const _Float16* prow = &sP[wib][t * 16 + l16][0];
            pa[t] = ldfrag(prow + hh * 8, prow + 16 + hh * 8);
        }

        // ---- V fragments for 32 keys, shared by both q-tiles ----
        const _Float16* vbase = vhead + kt + hh * 8;
        v16h vb[4];
        #pragma unroll
        for (int j = 0; j < 4; ++j) {
            const _Float16* pv = vbase + (size_t)(j * 16 + l16) * Lq;
            vb[j] = ldfrag(pv, pv + 16);
        }

        // ---- O += P(16x32) * V(32x64) for both q-tiles ----
        #pragma unroll
        for (int t = 0; t < 2; ++t)
            #pragma unroll
            for (int j = 0; j < 4; ++j)
                o[t][j] = wmma16(pa[t], vb[j], o[t][j]);
    }

    // ---- finalize and store O (f16 row-major, ready for proj GEMM) ----
    #pragma unroll
    for (int t = 0; t < 2; ++t)
        #pragma unroll
        for (int r = 0; r < 8; ++r) {
            float inv = 1.0f / lst[t][r];
            int m = qr0 + t * 16 + hh * 8 + r;
            _Float16* orow = oh + (size_t)m * DIM + h * HD;
            orow[l16]      = (_Float16)(o[t][0][r] * inv);
            orow[16 + l16] = (_Float16)(o[t][1][r] * inv);
            orow[32 + l16] = (_Float16)(o[t][2][r] * inv);
            orow[48 + l16] = (_Float16)(o[t][3][r] * inv);
        }
}

// ---------------------------------------------------------------------------
// Launch
// ---------------------------------------------------------------------------
extern "C" void kernel_launch(void* const* d_in, const int* in_sizes, int n_in,
                              void* d_out, int out_size, void* d_ws, size_t ws_size,
                              hipStream_t stream) {
    const float* x      = (const float*)d_in[0];
    const float* w_qkv  = (const float*)d_in[1];
    const float* b_qkv  = (const float*)d_in[2];
    const float* w_proj = (const float*)d_in[3];
    const float* b_proj = (const float*)d_in[4];
    const float* g_q    = (const float*)d_in[5];
    const float* g_k    = (const float*)d_in[6];
    float* out = (float*)d_out;

    // Carve workspace (256B aligned regions).
    char* p = (char*)d_ws;
    auto alloc = [&](size_t bytes) -> void* {
        void* r = (void*)p;
        p += (bytes + 255) & ~(size_t)255;
        return r;
    };
    _Float16* xh  = (_Float16*)alloc((size_t)Lq * DIM * 2);
    _Float16* wqh = (_Float16*)alloc((size_t)NQKV * DIM * 2);
    _Float16* wph = (_Float16*)alloc((size_t)DIM * DIM * 2);
    float*    qkv = (float*)   alloc((size_t)Lq * NQKV * 4);
    _Float16* qh  = (_Float16*)alloc((size_t)Lq * DIM * 2);
    _Float16* kh  = (_Float16*)alloc((size_t)Lq * DIM * 2);
    _Float16* vtb = (_Float16*)alloc((size_t)H * HD * Lq * 2);
    _Float16* oh  = (_Float16*)alloc((size_t)Lq * DIM * 2);

    // 1) convert operands to f16
    {
        int n0 = Lq * DIM, n1 = NQKV * DIM, n2 = DIM * DIM;
        cvt_f32_to_f16<<<(n0 + 255) / 256, 256, 0, stream>>>(x,      xh,  n0);
        cvt_f32_to_f16<<<(n1 + 255) / 256, 256, 0, stream>>>(w_qkv,  wqh, n1);
        cvt_f32_to_f16<<<(n2 + 255) / 256, 256, 0, stream>>>(w_proj, wph, n2);
    }

    // 2) QKV projection: [2048 x 4608] = xh @ wqh^T + b_qkv
    {
        int waves  = (Lq / 64) * (NQKV / 64);     // 32 * 72
        int blocks = (waves * 32) / 256;
        wmma_gemm_f16<<<blocks, 256, 0, stream>>>(xh, wqh, b_qkv, qkv, Lq, NQKV, DIM);
    }

    // 3) RMSNorm(q,k) + build f16 q/k and transposed V
    rmsnorm_qkv<<<(Lq * H) / 8, 256, 0, stream>>>(qkv, g_q, g_k, qh, kh, vtb);

    // 4) attention: one wave per (head, 32-query tile), 4 waves/block
    flash_attn<<<(H * (Lq / 32)) / 4, 128, 0, stream>>>(qh, kh, vtb, oh);

    // 5) output projection: [2048 x 1536] = oh @ wph^T + b_proj
    {
        int waves  = (Lq / 64) * (DIM / 64);      // 32 * 24
        int blocks = (waves * 32) / 256;
        wmma_gemm_f16<<<blocks, 256, 0, stream>>>(oh, wph, b_proj, out, Lq, DIM, DIM);
    }
}
